// Seq2SeqAttentionDecoder_86904368267438
// MI455X (gfx1250) — matrix operand order repaired and checked by reading, not compiled, not run on hardware
//
#include <hip/hip_runtime.h>
#include <hip/hip_bf16.h>

typedef __attribute__((ext_vector_type(16))) __bf16 v16bf;
typedef __attribute__((ext_vector_type(8)))  __bf16 v8bf;
typedef __attribute__((ext_vector_type(8)))  float  v8f;

#define H_DIM  512
#define HEADS  2
#define DH     256
#define B_DIM  64
#define S_DIM  1024
#define T_DIM  64
#define G_DIM  1536              // 3*H
#define BH     (B_DIM * H_DIM)  // 32768

// ---------------------------------------------------------------------------
// Tiled WMMA GEMM:  C[M, N] = A[M, K] @ W[N, K]^T   (bf16 in, f32 acc)
// One wave per block; each wave computes a 16 x (16*NTILES) strip of C,
// reusing the A fragment across NTILES WMMAs per K-step.
// grid.x = N/(16*NTILES), grid.y = M/16.
// Fragment layouts per CDNA5 ISA 7.12.2:
//   A (16x32, 16-bit): lane l holds row M=l&15; halfs 0..7 = K {koff..koff+7},
//                      halfs 8..15 = K {16+koff..16+koff+7}, koff=(l>>4)*8
//   B (32x16, 16-bit): lane l holds col N=l&15; same K order as A:
//                      halfs 0..7 = K {koff..koff+7}, halfs 8..15 = K {16+koff..16+koff+7}
//   C/D (16x16 f32):   lanes 0-15 N=l M=v; lanes 16-31 N=l-16 M=v+8
// ---------------------------------------------------------------------------
template<int KDIM, bool AVEC, int NTILES, bool OUT_BF>
__global__ void wmma_gemm_bf16(const __bf16* __restrict__ A, int lda,
                               const __bf16* __restrict__ W, int ldw,
                               float* __restrict__ Cf, __bf16* __restrict__ Cbf,
                               int ldc)
{
    const int lane = threadIdx.x;
    const int r    = lane & 15;
    const int koff = (lane >> 4) << 3;   // 0 or 8
    const int kbb  = (lane >> 4) << 4;   // 0 or 16

    const __bf16* arow = A + (size_t)(blockIdx.y * 16 + r) * lda;
    const __bf16* wrow[NTILES];
#pragma unroll
    for (int j = 0; j < NTILES; ++j)
        wrow[j] = W + (size_t)(blockIdx.x * (16 * NTILES) + j * 16 + r) * ldw;

    v8f acc[NTILES];
#pragma unroll
    for (int j = 0; j < NTILES; ++j) acc[j] = (v8f){};

#pragma unroll 4
    for (int kk = 0; kk < KDIM; kk += 32) {
        v16bf a;
        if (AVEC) {
            v8bf lo = *(const v8bf*)(arow + kk + koff);
            v8bf hi = *(const v8bf*)(arow + kk + 16 + koff);
            a = __builtin_shufflevector(lo, hi,
                    0, 1, 2, 3, 4, 5, 6, 7, 8, 9, 10, 11, 12, 13, 14, 15);
        } else {  // unaligned row stride (lda = 513): scalar gathers
#pragma unroll
            for (int i = 0; i < 8; ++i) {
                a[i]     = arow[kk + koff + i];
                a[i + 8] = arow[kk + 16 + koff + i];
            }
        }
#pragma unroll
        for (int j = 0; j < NTILES; ++j) {
            v16bf b = __builtin_shufflevector(*(const v8bf*)(wrow[j] + kk + koff), *(const v8bf*)(wrow[j] + kk + 16 + koff), 0, 1, 2, 3, 4, 5, 6, 7, 8, 9, 10, 11, 12, 13, 14, 15);
            acc[j] = __builtin_amdgcn_wmma_f32_16x16x32_bf16(
                         false, a, false, b, (short)0, acc[j], false, false);
        }
    }

    const int mb = blockIdx.y * 16 + ((lane >> 4) << 3);
#pragma unroll
    for (int j = 0; j < NTILES; ++j) {
        const int n = blockIdx.x * (16 * NTILES) + j * 16 + r;
#pragma unroll
        for (int v = 0; v < 8; ++v) {
            if (OUT_BF) Cbf[(size_t)(mb + v) * ldc + n] = (__bf16)acc[j][v];
            else        Cf [(size_t)(mb + v) * ldc + n] = acc[j][v];
        }
    }
}

// ---------------------------------------------------------------------------
// Fused K/V projection: Kbf/Vbf[(b*S+s), :] = enc[(b*S+s), :] @ Wk^T / Wv^T
// A fragment converted f32->bf16 on the fly; K and V accumulators share A.
// grid.x = (B*S)/16 = 4096, grid.y = H/16 = 32, 32 threads (1 wave).
// Output stored bf16 so all 134 MB of K+V state stays resident in 192 MB L2.
// ---------------------------------------------------------------------------
__global__ void kv_proj_kernel(const float* __restrict__ enc,
                               const __bf16* __restrict__ Wk,
                               const __bf16* __restrict__ Wv,
                               __bf16* __restrict__ Kout,
                               __bf16* __restrict__ Vout)
{
    const int lane = threadIdx.x;
    const int r    = lane & 15;
    const int koff = (lane >> 4) << 3;
    const int kbb  = (lane >> 4) << 4;

    const float*  arow = enc + ((size_t)blockIdx.x * 16 + r) * H_DIM;
    const __bf16* krow = Wk  + (size_t)(blockIdx.y * 16 + r) * H_DIM;
    const __bf16* vrow = Wv  + (size_t)(blockIdx.y * 16 + r) * H_DIM;

    v8f accK = {}, accV = {};
#pragma unroll 4
    for (int kk = 0; kk < H_DIM; kk += 32) {
        v8bf lo = __builtin_convertvector(*(const v8f*)(arow + kk + koff),      v8bf);
        v8bf hi = __builtin_convertvector(*(const v8f*)(arow + kk + 16 + koff), v8bf);
        v16bf a = __builtin_shufflevector(lo, hi,
                      0, 1, 2, 3, 4, 5, 6, 7, 8, 9, 10, 11, 12, 13, 14, 15);
        v16bf bk = __builtin_shufflevector(*(const v8bf*)(krow + kk + koff), *(const v8bf*)(krow + kk + 16 + koff), 0, 1, 2, 3, 4, 5, 6, 7, 8, 9, 10, 11, 12, 13, 14, 15);
        v16bf bv = __builtin_shufflevector(*(const v8bf*)(vrow + kk + koff), *(const v8bf*)(vrow + kk + 16 + koff), 0, 1, 2, 3, 4, 5, 6, 7, 8, 9, 10, 11, 12, 13, 14, 15);
        accK = __builtin_amdgcn_wmma_f32_16x16x32_bf16(false, a, false, bk, (short)0, accK, false, false);
        accV = __builtin_amdgcn_wmma_f32_16x16x32_bf16(false, a, false, bv, (short)0, accV, false, false);
    }

    const int    n  = blockIdx.y * 16 + r;
    const size_t mb = (size_t)blockIdx.x * 16 + ((lane >> 4) << 3);
#pragma unroll
    for (int v = 0; v < 8; ++v) {
        Kout[(mb + v) * H_DIM + n] = (__bf16)accK[v];
        Vout[(mb + v) * H_DIM + n] = (__bf16)accV[v];
    }
}

// ---------------------------------------------------------------------------
// Attention for one step: grid = B*HEADS = 128 blocks, 256 threads.
// scores = (q . K)*scale, softmax over S, ctx = attn @ V  -> bf16 ctx
// K/V are bf16 (L2 resident); math in f32.
// ---------------------------------------------------------------------------
__global__ void attn_kernel(const float* __restrict__ q,
                            const __bf16* __restrict__ Kbf,
                            const __bf16* __restrict__ Vbf,
                            __bf16* __restrict__ ctx_bf)
{
    __shared__ float qs[DH];
    __shared__ float sattn[S_DIM];
    __shared__ float red[256];

    const int tid  = threadIdx.x;
    const int b    = blockIdx.x >> 1;
    const int head = blockIdx.x & 1;

    if (tid < DH) qs[tid] = q[b * H_DIM + head * DH + tid];
    __syncthreads();

    float sc[4];
    float lmax = -1e30f;
#pragma unroll
    for (int i = 0; i < 4; ++i) {
        const int s = tid + i * 256;
        const __bf16* krow = Kbf + ((size_t)b * S_DIM + s) * H_DIM + head * DH;
        float acc = 0.f;
#pragma unroll 4
        for (int d = 0; d < DH; d += 8) {
            v8bf kv = *(const v8bf*)(krow + d);
#pragma unroll
            for (int j = 0; j < 8; ++j) acc += (float)kv[j] * qs[d + j];
        }
        sc[i] = acc * 0.0625f;              // 1/sqrt(256)
        lmax = fmaxf(lmax, sc[i]);
    }

    red[tid] = lmax; __syncthreads();
    for (int o = 128; o > 0; o >>= 1) {
        if (tid < o) red[tid] = fmaxf(red[tid], red[tid + o]);
        __syncthreads();
    }
    const float gmax = red[0]; __syncthreads();

    float lsum = 0.f;
#pragma unroll
    for (int i = 0; i < 4; ++i) {
        float e = __expf(sc[i] - gmax);
        sattn[tid + i * 256] = e;
        lsum += e;
    }
    red[tid] = lsum; __syncthreads();
    for (int o = 128; o > 0; o >>= 1) {
        if (tid < o) red[tid] += red[tid + o];
        __syncthreads();
    }
    const float inv = 1.0f / red[0];

    // ctx: thread tid owns output channel d = tid (coalesced V reads).
    float acc = 0.f;
    const __bf16* vbase = Vbf + (size_t)b * S_DIM * H_DIM + head * DH + tid;
#pragma unroll 8
    for (int s = 0; s < S_DIM; ++s)
        acc += sattn[s] * (float)vbase[(size_t)s * H_DIM];

    ctx_bf[b * H_DIM + head * DH + tid] = (__bf16)(acc * inv);
}

// ---------------------------------------------------------------------------
// GRU gates (elementwise). gi lacks the x*w_ih[:,512] rank-1 term and biases
// (Wo was folded into W_ihc) — added here. 128 blocks x 256 threads = B*H.
// ---------------------------------------------------------------------------
__device__ inline float sigmoidf_(float x) { return 1.0f / (1.0f + __expf(-x)); }

__global__ void gru_gates(const float* __restrict__ gi, const float* __restrict__ gh,
                          const float* __restrict__ b_ih, const float* __restrict__ b_hh,
                          const float* __restrict__ X, const float* __restrict__ w_ih,
                          float* __restrict__ hf, __bf16* __restrict__ hb,
                          float* __restrict__ Hall, int t)
{
    const int idx = blockIdx.x * 256 + threadIdx.x;   // 0..32767
    const int b = idx >> 9, j = idx & 511;
    const float x = X[b * T_DIM + t];

    const float* gib = gi + (size_t)b * G_DIM;
    const float* ghb = gh + (size_t)b * G_DIM;

    const float gr  = gib[j]        + x * w_ih[(size_t)(j)        * 513 + 512] + b_ih[j]
                    + ghb[j]        + b_hh[j];
    const float gz  = gib[512 + j]  + x * w_ih[(size_t)(512 + j)  * 513 + 512] + b_ih[512 + j]
                    + ghb[512 + j]  + b_hh[512 + j];
    const float gin = gib[1024 + j] + x * w_ih[(size_t)(1024 + j) * 513 + 512] + b_ih[1024 + j];
    const float ghn = ghb[1024 + j] + b_hh[1024 + j];

    const float r = sigmoidf_(gr);
    const float z = sigmoidf_(gz);
    const float n = tanhf(gin + r * ghn);
    const float hp = hf[idx];
    const float hn_ = (1.f - z) * n + z * hp;

    hf[idx] = hn_;
    hb[idx] = (__bf16)hn_;
    Hall[(size_t)t * BH + idx] = hn_;
}

// ---------------------------------------------------------------------------
// Small helpers
// ---------------------------------------------------------------------------
__global__ void cvt_f32_bf16(const float* __restrict__ src, __bf16* __restrict__ dst, int n)
{
    int i = blockIdx.x * 256 + threadIdx.x;
    if (i < n) dst[i] = (__bf16)src[i];
}

// WoT[n, i] = Wo[i, n]  (512x512), bf16 output
__global__ void transpose_wo(const float* __restrict__ src, __bf16* __restrict__ dst)
{
    int i = blockIdx.x * 256 + threadIdx.x;   // 262144
    int nrow = i >> 9, c = i & 511;
    dst[i] = (__bf16)src[c * 512 + nrow];
}

__global__ void init_h(const float* __restrict__ hs, float* __restrict__ hf,
                       __bf16* __restrict__ hb)
{
    int i = blockIdx.x * 256 + threadIdx.x;
    if (i < BH) { float v = hs[i]; hf[i] = v; hb[i] = (__bf16)v; }
}

// outputs[b, t] = H_all[t, b, :] . Wd + bd    (4096 outputs)
__global__ void final_out(const float* __restrict__ Hall, const float* __restrict__ Wd,
                          const float* __restrict__ bd, float* __restrict__ out)
{
    int idx = blockIdx.x * 256 + threadIdx.x;   // 0..4095
    int b = idx >> 6, t = idx & 63;
    const float* hrow = Hall + (size_t)t * BH + (size_t)b * H_DIM;
    float acc = bd[0];
#pragma unroll 8
    for (int j = 0; j < H_DIM; ++j) acc += hrow[j] * Wd[j];
    out[b * T_DIM + t] = acc;
}

__global__ void copy_hfinal(const float* __restrict__ Hall, float* __restrict__ dst)
{
    int i = blockIdx.x * 256 + threadIdx.x;
    if (i < BH) dst[i] = Hall[(size_t)(T_DIM - 1) * BH + i];
}

// ---------------------------------------------------------------------------
extern "C" void kernel_launch(void* const* d_in, const int* in_sizes, int n_in,
                              void* d_out, int out_size, void* d_ws, size_t ws_size,
                              hipStream_t stream)
{
    const float* X    = (const float*)d_in[0];
    const float* enc  = (const float*)d_in[1];
    const float* hs   = (const float*)d_in[2];
    const float* Wq   = (const float*)d_in[3];
    const float* Wk   = (const float*)d_in[4];
    const float* Wv   = (const float*)d_in[5];
    const float* Wo   = (const float*)d_in[6];
    const float* w_ih = (const float*)d_in[7];
    const float* w_hh = (const float*)d_in[8];
    const float* b_ih = (const float*)d_in[9];
    const float* b_hh = (const float*)d_in[10];
    const float* Wd   = (const float*)d_in[11];
    const float* bd   = (const float*)d_in[12];
    float* out = (float*)d_out;

    // ---- workspace layout (bytes) ----
    char* base = (char*)d_ws;
    size_t off = 0;
    auto nxt = [&](size_t bytes) { size_t o = off; off += (bytes + 255) & ~(size_t)255; return o; };

    __bf16* Kbf     = (__bf16*)(base + nxt((size_t)B_DIM * S_DIM * H_DIM * 2));
    __bf16* Vbf     = (__bf16*)(base + nxt((size_t)B_DIM * S_DIM * H_DIM * 2));
    __bf16* Wq_bf   = (__bf16*)(base + nxt((size_t)H_DIM * H_DIM * 2));
    __bf16* Wk_bf   = (__bf16*)(base + nxt((size_t)H_DIM * H_DIM * 2));
    __bf16* Wv_bf   = (__bf16*)(base + nxt((size_t)H_DIM * H_DIM * 2));
    __bf16* WoT_bf  = (__bf16*)(base + nxt((size_t)H_DIM * H_DIM * 2));
    __bf16* whh_bf  = (__bf16*)(base + nxt((size_t)G_DIM * H_DIM * 2));
    __bf16* wih_bf  = (__bf16*)(base + nxt((size_t)G_DIM * 513 * 2));
    __bf16* Wihc_bf = (__bf16*)(base + nxt((size_t)G_DIM * H_DIM * 2));   // w_ih[:, :512] @ Wo
    float*  qbuf    = (float*) (base + nxt((size_t)B_DIM * H_DIM * 4));
    __bf16* ctx_bf  = (__bf16*)(base + nxt((size_t)B_DIM * H_DIM * 2));
    float*  gi      = (float*) (base + nxt((size_t)B_DIM * G_DIM * 4));
    float*  gh      = (float*) (base + nxt((size_t)B_DIM * G_DIM * 4));
    float*  hf      = (float*) (base + nxt((size_t)BH * 4));
    __bf16* hb      = (__bf16*)(base + nxt((size_t)BH * 2));
    float*  Hall    = (float*) (base + nxt((size_t)T_DIM * BH * 4));
    (void)ws_size; (void)in_sizes; (void)n_in; (void)out_size;

    // ---- passthrough enc_outputs into d_out (independent, start early) ----
    hipMemcpyAsync(out + B_DIM * T_DIM, enc,
                   (size_t)B_DIM * S_DIM * H_DIM * sizeof(float),
                   hipMemcpyDeviceToDevice, stream);

    // ---- setup: weight conversion + Wo fold + K/V projection ----
    cvt_f32_bf16<<<(H_DIM * H_DIM + 255) / 256, 256, 0, stream>>>(Wq, Wq_bf, H_DIM * H_DIM);
    cvt_f32_bf16<<<(H_DIM * H_DIM + 255) / 256, 256, 0, stream>>>(Wk, Wk_bf, H_DIM * H_DIM);
    cvt_f32_bf16<<<(H_DIM * H_DIM + 255) / 256, 256, 0, stream>>>(Wv, Wv_bf, H_DIM * H_DIM);
    cvt_f32_bf16<<<(G_DIM * H_DIM + 255) / 256, 256, 0, stream>>>(w_hh, whh_bf, G_DIM * H_DIM);
    cvt_f32_bf16<<<(G_DIM * 513 + 255) / 256, 256, 0, stream>>>(w_ih, wih_bf, G_DIM * 513);
    transpose_wo<<<(H_DIM * H_DIM + 255) / 256, 256, 0, stream>>>(Wo, WoT_bf);

    // W_ihc[1536,512] = w_ih[:, :512] @ Wo   (lda=513 -> scalar A path), bf16 out
    wmma_gemm_bf16<H_DIM, false, 4, true>
        <<<dim3(H_DIM / 64, G_DIM / 16), 32, 0, stream>>>(
            wih_bf, 513, WoT_bf, H_DIM, nullptr, Wihc_bf, H_DIM);

    // K/V projection (bf16 result -> L2 resident for the whole scan)
    kv_proj_kernel<<<dim3((B_DIM * S_DIM) / 16, H_DIM / 16), 32, 0, stream>>>(
        enc, Wk_bf, Wv_bf, Kbf, Vbf);

    init_h<<<(BH + 255) / 256, 256, 0, stream>>>(hs, hf, hb);

    // ---- sequential decoder scan ----
    for (int t = 0; t < T_DIM; ++t) {
        // q = h @ Wq^T   (64x512x512)
        wmma_gemm_bf16<H_DIM, true, 4, false>
            <<<dim3(H_DIM / 64, B_DIM / 16), 32, 0, stream>>>(
                hb, H_DIM, Wq_bf, H_DIM, qbuf, nullptr, H_DIM);

        // attention -> ctx (bf16)
        attn_kernel<<<B_DIM * HEADS, 256, 0, stream>>>(qbuf, Kbf, Vbf, ctx_bf);

        // gi = ctx @ W_ihc^T   (64x1536x512)
        wmma_gemm_bf16<H_DIM, true, 4, false>
            <<<dim3(G_DIM / 64, B_DIM / 16), 32, 0, stream>>>(
                ctx_bf, H_DIM, Wihc_bf, H_DIM, gi, nullptr, G_DIM);

        // gh = h @ w_hh^T      (64x1536x512)
        wmma_gemm_bf16<H_DIM, true, 4, false>
            <<<dim3(G_DIM / 64, B_DIM / 16), 32, 0, stream>>>(
                hb, H_DIM, whh_bf, H_DIM, gh, nullptr, G_DIM);

        // GRU elementwise update
        gru_gates<<<BH / 256, 256, 0, stream>>>(gi, gh, b_ih, b_hh, X, w_ih,
                                                hf, hb, Hall, t);
    }

    // ---- outputs ----
    final_out<<<(B_DIM * T_DIM) / 256, 256, 0, stream>>>(Hall, Wd, bd, out);
    copy_hfinal<<<(BH + 255) / 256, 256, 0, stream>>>(
        Hall, out + B_DIM * T_DIM + (size_t)B_DIM * S_DIM * H_DIM);
}
